// GAT_56590489092397
// MI455X (gfx1250) — matrix operand order repaired and checked by reading, not compiled, not run on hardware
//
#include <hip/hip_runtime.h>
#include <hip/hip_bf16.h>

typedef __attribute__((ext_vector_type(16))) _Float16 v16h;
typedef __attribute__((ext_vector_type(8)))  float    v8f;

#define NNODES 50000
#define NEDGES 800000
#define ETOT   850000       // NEDGES + NNODES self loops
#define KDIM   128
#define GRAPHS 64
#define OUTD   64
#define NEGS   0.2f

// ---------------- utility kernels ----------------

__global__ void fill_f32(float* __restrict__ p, float v, int n) {
    int i = blockIdx.x * blockDim.x + threadIdx.x;
    if (i < n) p[i] = v;
}

__global__ void cvt_f16(const float* __restrict__ src, _Float16* __restrict__ dst, int n) {
    int i = blockIdx.x * blockDim.x + threadIdx.x;
    if (i < n) dst[i] = (_Float16)src[i];
}

// Convert + transpose weights: W [128][hc] f32 row-major -> WT [hc][128] f16 row-major.
__global__ void cvt_w_t(const float* __restrict__ w, _Float16* __restrict__ wt, int hc) {
    int i = blockIdx.x * blockDim.x + threadIdx.x;
    if (i >= KDIM * hc) return;
    int k = i / hc, n = i - k * hc;
    wt[n * KDIM + k] = (_Float16)w[i];
}

// ---------------- WMMA dual GEMM: XL = X*Wl, XR = X*Wr ----------------
// X: [NNODES][128] f16 row-major.  WT: [hc][128] f16 (transposed weights).
// One wave (32 threads) computes one 16x16 tile of both outputs.
// grid.x = column tiles (so consecutive blocks reuse the same A tile), grid.y = row tiles.
__global__ void __launch_bounds__(32)
gemm_dual_wmma(const _Float16* __restrict__ x16,
               const _Float16* __restrict__ wlT,
               const _Float16* __restrict__ wrT,
               float* __restrict__ XL, float* __restrict__ XR, int hc)
{
    __shared__ __align__(16) _Float16 As[16 * KDIM];   // 4 KB tile
    const int lane = threadIdx.x;
    const int n0   = blockIdx.x * 16;                  // output-col tile
    const int r0   = blockIdx.y * 16;                  // node-row tile

    // cooperative 16x128 f16 tile load: 256 x uint4 (16B), 8 per lane
    const uint4* gsrc = (const uint4*)(x16 + r0 * KDIM);
    uint4* lds = (uint4*)As;
    #pragma unroll
    for (int j = 0; j < 8; ++j) lds[lane + 32 * j] = gsrc[lane + 32 * j];
    __syncthreads();

    const int half = lane >> 4;        // A: K sub-block select; B: K-half select
    const int m    = lane & 15;        // A row
    const int nc   = lane & 15;        // B/C column within tile
    const int ncol = n0 + nc;

    // per-lane base of this column's transposed weight row (contiguous in K)
    const int wbase = ncol * KDIM;

    v8f cl = {}; v8f cr = {};
    #pragma unroll
    for (int kt = 0; kt < KDIM / 32; ++kt) {
        const int k0 = kt * 32;
        // A fragment 16x32: lanes 0-15 -> K 0..7 & 16..23 ; lanes 16-31 -> K 8..15 & 24..31
        union Frag { uint4 u4[2]; v16h v; };
        Frag af;
        af.u4[0] = *(const uint4*)&As[m * KDIM + k0 + half * 8];
        af.u4[1] = *(const uint4*)&As[m * KDIM + k0 + 16 + half * 8];
        // B fragment 32x16: lane holds column ncol, K = k0 + half*8 + [0..7] & k0 + 16 + half*8 + [0..7]
        Frag bfl, bfr;
        const int kb = wbase + k0 + half * 8;
        bfl.u4[0] = *(const uint4*)&wlT[kb];
        bfl.u4[1] = *(const uint4*)&wlT[kb + 16];
        bfr.u4[0] = *(const uint4*)&wrT[kb];
        bfr.u4[1] = *(const uint4*)&wrT[kb + 16];
        cl = __builtin_amdgcn_wmma_f32_16x16x32_f16(false, af.v, false, bfl.v, (short)0, cl, false, false);
        cr = __builtin_amdgcn_wmma_f32_16x16x32_f16(false, af.v, false, bfr.v, (short)0, cr, false, false);
    }
    // C/D layout: VGPR v -> row v (lanes 0-15) / v+8 (lanes 16-31), col = lane&15
    #pragma unroll
    for (int v = 0; v < 8; ++v) {
        const int row = r0 + v + 8 * half;
        XL[row * hc + ncol] = cl[v];
        XR[row * hc + ncol] = cr[v];
    }
}

// ---------------- edge kernels (templated on layer shape) ----------------

__device__ __forceinline__ void atomicMaxF(float* addr, float val) {
    // sign-aware integer-ordered float max (monotone under both orderings)
    if (val >= 0.0f) atomicMax((int*)addr, __float_as_int(val));
    else             atomicMin((unsigned int*)addr, __float_as_uint(val));
}

__device__ __forceinline__ void edge_st(const int* __restrict__ ei, int e, int& s, int& t) {
    if (e < NEDGES) { s = ei[e]; t = ei[NEDGES + e]; }
    else            { s = e - NEDGES; t = s; }        // self loops
}

template <int HEADS, int CHEAD>
__global__ void edge_score(const int* __restrict__ ei,
                           const float* __restrict__ XL, const float* __restrict__ XR,
                           const float* __restrict__ att,
                           float* __restrict__ score, float* __restrict__ smax)
{
    const int tid = blockIdx.x * blockDim.x + threadIdx.x;
    if (tid >= ETOT * HEADS) return;
    const int e = tid / HEADS, h = tid - e * HEADS;
    int s, t; edge_st(ei, e, s, t);
    constexpr int HC = HEADS * CHEAD;
    const float* pl = XL + s * HC + h * CHEAD;
    const float* pr = XR + t * HC + h * CHEAD;
    const float* pa = att + h * CHEAD;
    float acc = 0.0f;
    #pragma unroll
    for (int c = 0; c < CHEAD; c += 4) {
        float4 a = *(const float4*)(pl + c);
        float4 b = *(const float4*)(pr + c);
        float4 w = *(const float4*)(pa + c);
        float g0 = a.x + b.x; g0 = g0 > 0.f ? g0 : NEGS * g0;
        float g1 = a.y + b.y; g1 = g1 > 0.f ? g1 : NEGS * g1;
        float g2 = a.z + b.z; g2 = g2 > 0.f ? g2 : NEGS * g2;
        float g3 = a.w + b.w; g3 = g3 > 0.f ? g3 : NEGS * g3;
        acc += g0 * w.x + g1 * w.y + g2 * w.z + g3 * w.w;
    }
    score[tid] = acc;
    atomicMaxF(&smax[t * HEADS + h], acc);
}

template <int HEADS>
__global__ void edge_exp(const int* __restrict__ ei,
                         float* __restrict__ score, const float* __restrict__ smax,
                         float* __restrict__ den)
{
    const int tid = blockIdx.x * blockDim.x + threadIdx.x;
    if (tid >= ETOT * HEADS) return;
    const int e = tid / HEADS, h = tid - e * HEADS;
    int s, t; edge_st(ei, e, s, t); (void)s;
    const float ex = __expf(score[tid] - smax[t * HEADS + h]);
    score[tid] = ex;
    atomicAdd(&den[t * HEADS + h], ex);
}

template <int HEADS, int CHEAD>
__global__ void edge_scatter(const int* __restrict__ ei,
                             const float* __restrict__ score, const float* __restrict__ den,
                             const float* __restrict__ XL, float* __restrict__ acc)
{
    const int tid = blockIdx.x * blockDim.x + threadIdx.x;
    if (tid >= ETOT * HEADS) return;
    const int e = tid / HEADS, h = tid - e * HEADS;
    int s, t; edge_st(ei, e, s, t);
    constexpr int HC = HEADS * CHEAD;
    const float alpha = score[tid] / den[t * HEADS + h];
    const float* pl = XL + s * HC + h * CHEAD;
    float* pd = acc + t * HC + h * CHEAD;
    #pragma unroll
    for (int c = 0; c < CHEAD; c += 4) {
        float4 a = *(const float4*)(pl + c);
        atomicAdd(pd + c + 0, alpha * a.x);
        atomicAdd(pd + c + 1, alpha * a.y);
        atomicAdd(pd + c + 2, alpha * a.z);
        atomicAdd(pd + c + 3, alpha * a.w);
    }
}

// ---------------- finalize + pooling ----------------

__global__ void finalize_mid(const float* __restrict__ acc, const float* __restrict__ bias,
                             _Float16* __restrict__ xnext, int hc)
{
    const int i = blockIdx.x * blockDim.x + threadIdx.x;
    if (i >= NNODES * hc) return;
    const int j = i % hc;
    float v = acc[i] + bias[j];
    v = v > 0.f ? v : NEGS * v;            // outer leaky_relu after layer
    xnext[i] = (_Float16)v;
}

__global__ void finalize_last(float* __restrict__ acc, const float* __restrict__ bias)
{
    const int i = blockIdx.x * blockDim.x + threadIdx.x;
    if (i >= NNODES * OUTD) return;
    const int j = i % OUTD;
    float v = acc[i] + bias[j];            // heads==1 so mean over heads is identity
    acc[i] = v > 0.f ? v : NEGS * v;
}

__global__ void pool_acc(const float* __restrict__ xf, const int* __restrict__ batch,
                         float* __restrict__ gsum, float* __restrict__ gcnt)
{
    const int i = blockIdx.x * blockDim.x + threadIdx.x;
    if (i >= NNODES * OUTD) return;
    const int n = i / OUTD, j = i - n * OUTD;
    const int g = batch[n];
    atomicAdd(&gsum[g * OUTD + j], xf[i]);
    if (j == 0) atomicAdd(&gcnt[g], 1.0f);
}

__global__ void pool_out(const float* __restrict__ gsum, const float* __restrict__ gcnt,
                         float* __restrict__ out)
{
    const int i = blockIdx.x * blockDim.x + threadIdx.x;
    if (i >= GRAPHS * OUTD) return;
    out[i] = gsum[i] / fmaxf(gcnt[i / OUTD], 1.0f);
}

// ---------------- host orchestration ----------------

static inline int cdiv(int a, int b) { return (a + b - 1) / b; }

extern "C" void kernel_launch(void* const* d_in, const int* in_sizes, int n_in,
                              void* d_out, int out_size, void* d_ws, size_t ws_size,
                              hipStream_t stream)
{
    const float* x     = (const float*)d_in[0];
    const int*   ei    = (const int*)d_in[1];
    const int*   batch = (const int*)d_in[2];
    const float* Wl[3]  = {(const float*)d_in[3],  (const float*)d_in[7],  (const float*)d_in[11]};
    const float* Wr[3]  = {(const float*)d_in[4],  (const float*)d_in[8],  (const float*)d_in[12]};
    const float* att[3] = {(const float*)d_in[5],  (const float*)d_in[9],  (const float*)d_in[13]};
    const float* bia[3] = {(const float*)d_in[6],  (const float*)d_in[10], (const float*)d_in[14]};

    char* ws = (char*)d_ws;
    size_t off = 0;
    auto alloc = [&](size_t bytes) -> char* {
        char* p = ws + off;
        off = (off + bytes + 255) & ~(size_t)255;
        return p;
    };

    float*    XLp  = (float*)alloc((size_t)NNODES * KDIM * sizeof(float));   // 25.6 MB
    float*    XRp  = (float*)alloc((size_t)NNODES * KDIM * sizeof(float));   // 25.6 MB (aliased as ACC)
    float*    ACC  = XRp;                                                    // XR dead after edge_score
    float*    SC   = (float*)alloc((size_t)ETOT * 4 * sizeof(float));        // 13.6 MB
    float*    SMAX = (float*)alloc((size_t)NNODES * 4 * sizeof(float));
    float*    DEN  = (float*)alloc((size_t)NNODES * 4 * sizeof(float));
    _Float16* X16  = (_Float16*)alloc((size_t)NNODES * KDIM * sizeof(_Float16)); // 12.8 MB
    _Float16* WTL  = (_Float16*)alloc((size_t)KDIM * KDIM * sizeof(_Float16));   // transposed f16 weights
    _Float16* WTR  = (_Float16*)alloc((size_t)KDIM * KDIM * sizeof(_Float16));
    float*    GS   = (float*)alloc((size_t)(GRAPHS * OUTD + GRAPHS) * sizeof(float));

    const int TB = 256;
    cvt_f16<<<cdiv(NNODES * KDIM, TB), TB, 0, stream>>>(x, X16, NNODES * KDIM);

    for (int li = 0; li < 3; ++li) {
        const int h  = (li < 2) ? 4 : 1;
        const int hc = (li < 2) ? 128 : 64;

        cvt_w_t<<<cdiv(KDIM * hc, TB), TB, 0, stream>>>(Wl[li], WTL, hc);
        cvt_w_t<<<cdiv(KDIM * hc, TB), TB, 0, stream>>>(Wr[li], WTR, hc);

        dim3 gg(hc / 16, NNODES / 16);
        gemm_dual_wmma<<<gg, 32, 0, stream>>>(X16, WTL, WTR, XLp, XRp, hc);

        fill_f32<<<cdiv(NNODES * h, TB), TB, 0, stream>>>(SMAX, -3.0e38f, NNODES * h);

        const int tot = ETOT * h;
        if (li < 2)
            edge_score<4, 32><<<cdiv(tot, TB), TB, 0, stream>>>(ei, XLp, XRp, att[li], SC, SMAX);
        else
            edge_score<1, 64><<<cdiv(tot, TB), TB, 0, stream>>>(ei, XLp, XRp, att[li], SC, SMAX);

        // XR no longer needed -> reuse as ACC
        fill_f32<<<cdiv(NNODES * h, TB), TB, 0, stream>>>(DEN, 0.0f, NNODES * h);
        fill_f32<<<cdiv(NNODES * hc, TB), TB, 0, stream>>>(ACC, 0.0f, NNODES * hc);

        if (li < 2) {
            edge_exp<4><<<cdiv(tot, TB), TB, 0, stream>>>(ei, SC, SMAX, DEN);
            edge_scatter<4, 32><<<cdiv(tot, TB), TB, 0, stream>>>(ei, SC, DEN, XLp, ACC);
            finalize_mid<<<cdiv(NNODES * hc, TB), TB, 0, stream>>>(ACC, bia[li], X16, hc);
        } else {
            edge_exp<1><<<cdiv(tot, TB), TB, 0, stream>>>(ei, SC, SMAX, DEN);
            edge_scatter<1, 64><<<cdiv(tot, TB), TB, 0, stream>>>(ei, SC, DEN, XLp, ACC);
            finalize_last<<<cdiv(NNODES * OUTD, TB), TB, 0, stream>>>(ACC, bia[li]);
        }
    }

    fill_f32<<<cdiv(GRAPHS * OUTD + GRAPHS, TB), TB, 0, stream>>>(GS, 0.0f, GRAPHS * OUTD + GRAPHS);
    pool_acc<<<cdiv(NNODES * OUTD, TB), TB, 0, stream>>>(ACC, batch, GS, GS + GRAPHS * OUTD);
    pool_out<<<cdiv(GRAPHS * OUTD, TB), TB, 0, stream>>>(GS, GS + GRAPHS * OUTD, (float*)d_out);
}
